// BottleneckIndependent_70222715289974
// MI455X (gfx1250) — matrix operand order repaired and checked by reading, not compiled, not run on hardware
//
#include <hip/hip_runtime.h>
#include <hip/hip_bf16.h>
#include <math.h>

#define BATCH 32
#define INPC 1024
#define WIDTHC 256
#define OUTPC 1024
#define NEXP 8
#define HH 14
#define PIX 196
#define EPS_BN 1e-5f

#define TILE_M 128      // 4 waves x 32 rows
#define TILE_N 64
#define TILE_K 32
#define LDS_STRIDE 40   // f16; 80B pitch == TDM pad (16 DW interval + 4 DW pad)
#define BLOCK 128       // 4 waves (wave32)

typedef _Float16 v8h  __attribute__((ext_vector_type(8)));
typedef _Float16 v16h __attribute__((ext_vector_type(16)));
typedef float    v8f  __attribute__((ext_vector_type(8)));
typedef unsigned int u32x4 __attribute__((ext_vector_type(4)));
typedef int          i32x4 __attribute__((ext_vector_type(4)));
typedef int          i32x8 __attribute__((ext_vector_type(8)));

// ---- WMMA fragment loads, per CDNA5 ISA 7.12.2 layouts -------------------
__device__ __forceinline__ v16h load_a_frag(const _Float16* As, int mtile, int lane) {
  const int row = mtile * 16 + (lane & 15);
  const int kg  = (lane >> 4) * 8;
  v8h lo = *(const v8h*)&As[row * LDS_STRIDE + kg];
  v8h hi = *(const v8h*)&As[row * LDS_STRIDE + kg + 16];
  return __builtin_shufflevector(lo, hi, 0,1,2,3,4,5,6,7,8,9,10,11,12,13,14,15);
}
__device__ __forceinline__ v16h load_b_frag(const _Float16* Bs, int ntile, int lane) {
  const int col = ntile * 16 + (lane & 15);
  const int ks  = (lane >> 4) * 8;
  v8h lo = *(const v8h*)&Bs[col * LDS_STRIDE + ks];
  v8h hi = *(const v8h*)&Bs[col * LDS_STRIDE + ks + 16];
  return __builtin_shufflevector(lo, hi, 0,1,2,3,4,5,6,7,8,9,10,11,12,13,14,15);
}

// ---- A-tile staging: TDM (tensor_load_to_lds) with cooperative fallback ---
// [128 rows x 32 k] f16 tile from cw (row pitch Ktot) -> As, stride-40 rows
// via TDM LDS padding (pad_interval=3: 16DW, pad_amount=3: 4DW).
__device__ __forceinline__ void stage_A_issue(const _Float16* cw_mbase, int Ktot, int kb,
                                              _Float16* As, int tid, int wave) {
#if __has_builtin(__builtin_amdgcn_tensor_load_to_lds)
  if (wave == 0) {
    const unsigned lds = (unsigned)(size_t)As;   // low 32 bits of generic = LDS offset
    const unsigned long long ga = (unsigned long long)(size_t)(cw_mbase + kb);
    u32x4 g0;
    g0[0] = 1u;                                  // count=1, user descriptor
    g0[1] = lds;                                 // lds_addr
    g0[2] = (unsigned)ga;                        // global_addr[31:0]
    g0[3] = (unsigned)((ga >> 32) & 0x1FFFFFFu) | (2u << 30);  // ga[56:32] | type=2
    i32x8 g1;
    g1[0] = (int)((1u << 16) | (1u << 20) | (3u << 22) | (3u << 25)); // 2B elems, pad en
    g1[1] = (int)(((unsigned)Ktot & 0xFFFFu) << 16);     // tensor_dim0[15:0]
    g1[2] = (int)(((unsigned)Ktot >> 16) | ((unsigned)TILE_M << 16)); // dim0 hi | tensor_dim1 lo
    g1[3] = (int)(32u << 16);                    // tile_dim0 = 32
    g1[4] = (int)(unsigned)TILE_M;               // tile_dim1 = 128
    g1[5] = (int)(unsigned)Ktot;                 // tensor_dim0_stride lo32
    g1[6] = 0; g1[7] = 0;
    i32x4 gz = {0, 0, 0, 0};
#if defined(__clang_major__) && (__clang_major__ >= 23)
    i32x8 gz8 = {0, 0, 0, 0, 0, 0, 0, 0};
    __builtin_amdgcn_tensor_load_to_lds(g0, g1, gz, gz, gz8, 0);
#else
    __builtin_amdgcn_tensor_load_to_lds(g0, g1, gz, gz, 0);
#endif
  }
#else
  const int row = tid;   // 128 threads, one 32-f16 row each
  const v8h* src = (const v8h*)(cw_mbase + (size_t)row * Ktot + kb);
  *(v8h*)&As[row * LDS_STRIDE]     = src[0];
  *(v8h*)&As[row * LDS_STRIDE + 8] = src[1];
#endif
}
__device__ __forceinline__ void stage_A_wait(int wave) {
#if __has_builtin(__builtin_amdgcn_tensor_load_to_lds)
  if (wave == 0) __builtin_amdgcn_s_wait_tensorcnt(0);
#endif
}

// ---- B-tile staging: thread owns one column (fixed pixel, 16 consecutive k).
// Unconditional clamped loads (coalesced across lanes), single guard flag,
// packed f16 conversion, two ds_store_b128 per thread.
__device__ __forceinline__ void stage_B_cols(const float* __restrict__ srcbase, // &in[kstart*PIX]
                                             int off, bool ok, int n, int kseg,
                                             _Float16* Bs) {
  const float* s = srcbase + (size_t)kseg * PIX + off;
  float v[16];
  #pragma unroll
  for (int j = 0; j < 16; ++j) v[j] = s[(size_t)j * PIX];
  #pragma unroll
  for (int j = 0; j < 16; ++j) v[j] = ok ? v[j] : 0.f;
  v8h h0, h1;
  #pragma unroll
  for (int j = 0; j < 8; ++j) { h0[j] = (_Float16)v[j]; h1[j] = (_Float16)v[j + 8]; }
  _Float16* d = &Bs[n * LDS_STRIDE + kseg];
  *(v8h*)d       = h0;
  *(v8h*)(d + 8) = h1;
}

// ---- routing --------------------------------------------------------------
__global__ void pool_kernel(const float* __restrict__ in, float* __restrict__ pooled, int C) {
  int idx = blockIdx.x * blockDim.x + threadIdx.x;
  if (idx >= BATCH * C) return;
  const float* p = in + (size_t)idx * PIX;
  float s = 0.f;
  for (int i = 0; i < PIX; ++i) s += p[i];
  pooled[idx] = s * (1.0f / PIX);
}

__global__ void route_kernel(const float* __restrict__ pooled, const float* __restrict__ w,
                             const float* __restrict__ bias, float* __restrict__ rw, int C) {
  int t = threadIdx.x;              // 256 = 32 samples x 8 experts
  int b = t >> 3, e = t & 7;
  const float* pb = pooled + (size_t)b * C;
  float s = bias[e];
  for (int c = 0; c < C; ++c) s += pb[c] * w[c * NEXP + e];
  rw[t] = 1.0f / (1.0f + expf(-s));
}

// ---- expert-weight combine (hoisted out of GEMM hot loop) -----------------
__global__ void combine13_kernel(const float* __restrict__ w, const float* __restrict__ rw,
                                 _Float16* __restrict__ cw, int M, int K) {
  const int b = blockIdx.y;
  const int chunk = blockIdx.x * blockDim.x + threadIdx.x;
  const int m  = (chunk * 8) / K;
  const int k0 = (chunk * 8) % K;
  float rwl[NEXP];
  #pragma unroll
  for (int e = 0; e < NEXP; ++e) rwl[e] = rw[b * NEXP + e];
  const float* base = w + (size_t)m * K + k0;
  float v[8] = {0.f, 0.f, 0.f, 0.f, 0.f, 0.f, 0.f, 0.f};
  #pragma unroll
  for (int e = 0; e < NEXP; ++e) {
    const float* we = base + (size_t)e * M * K;
    #pragma unroll
    for (int j = 0; j < 8; ++j) v[j] += rwl[e] * we[j];
  }
  v8h o;
  #pragma unroll
  for (int j = 0; j < 8; ++j) o[j] = (_Float16)v[j];
  *(v8h*)&cw[((size_t)b * M + m) * K + k0] = o;
}

__global__ void combine2_kernel(const float* __restrict__ w, const float* __restrict__ rw,
                                _Float16* __restrict__ cw) {
  const int KT = WIDTHC * 9;
  const int b = blockIdx.y;
  const int chunk = blockIdx.x * blockDim.x + threadIdx.x;
  const int o   = (chunk * 8) / KT;
  const int kk  = (chunk * 8) % KT;
  const int tap = kk / WIDTHC, i0 = kk % WIDTHC;
  float rwl[NEXP];
  #pragma unroll
  for (int e = 0; e < NEXP; ++e) rwl[e] = rw[b * NEXP + e];
  float v[8] = {0.f, 0.f, 0.f, 0.f, 0.f, 0.f, 0.f, 0.f};
  #pragma unroll
  for (int e = 0; e < NEXP; ++e) {
    const float* we = w + (((size_t)e * WIDTHC + o) * WIDTHC + i0) * 9 + tap;
    #pragma unroll
    for (int j = 0; j < 8; ++j) v[j] += rwl[e] * we[j * 9];
  }
  v8h out;
  #pragma unroll
  for (int j = 0; j < 8; ++j) out[j] = (_Float16)v[j];
  *(v8h*)&cw[((size_t)b * WIDTHC + o) * KT + kk] = out;
}

// ---- epilogue helper ------------------------------------------------------
__device__ __forceinline__ void store_bn(v8f acc[2][4], int obase, int pbase, int wave, int lane,
                                         const float* g, const float* bb,
                                         const float* mu, const float* var,
                                         const float* resid, float* ob) {
  #pragma unroll
  for (int ms = 0; ms < 2; ++ms) {
    #pragma unroll
    for (int n = 0; n < 4; ++n) {
      #pragma unroll
      for (int r = 0; r < 8; ++r) {
        const int o = obase + wave * 32 + ms * 16 + r + ((lane >> 4) << 3);
        const int p = pbase + n * 16 + (lane & 15);
        if (p < PIX) {
          const float inv = g[o] * rsqrtf(var[o] + EPS_BN);
          float v = acc[ms][n][r] * inv + (bb[o] - mu[o] * inv);
          if (resid) v += resid[(size_t)o * PIX + p];
          ob[(size_t)o * PIX + p] = fmaxf(v, 0.f);
        }
      }
    }
  }
}

// ---- stage 1 & 3 share a 1x1 GEMM body ------------------------------------
__device__ __forceinline__ void gemm1x1_body(const float* __restrict__ in, int Ktot,
                                             const _Float16* __restrict__ cwm,
                                             _Float16 (*As)[TILE_M * LDS_STRIDE],
                                             _Float16 (*Bs)[TILE_N * LDS_STRIDE],
                                             v8f acc[2][4],
                                             int tid, int wave, int lane, int pbase) {
  const int n = tid & 63, kseg = (tid >> 6) * 16;   // column-owner mapping
  const int p = pbase + n;
  const bool ok = p < PIX;
  const int off = ok ? p : (PIX - 1);
  const int NIT = Ktot / TILE_K;

  stage_A_issue(cwm, Ktot, 0, As[0], tid, wave);
  stage_B_cols(in, off, ok, n, kseg, Bs[0]);

  for (int it = 0; it < NIT; ++it) {
    const int cur = it & 1;
    stage_A_wait(wave);
    __syncthreads();
    if (it + 1 < NIT) {   // pipeline: next tile overlaps this tile's WMMAs
      const int kn = (it + 1) * TILE_K;
      stage_A_issue(cwm, Ktot, kn, As[cur ^ 1], tid, wave);
      stage_B_cols(in + (size_t)kn * PIX, off, ok, n, kseg, Bs[cur ^ 1]);
    }
    #pragma unroll
    for (int ms = 0; ms < 2; ++ms) {
      v16h a = load_a_frag(As[cur], wave * 2 + ms, lane);
      #pragma unroll
      for (int nn = 0; nn < 4; ++nn) {
        v16h bf = load_b_frag(Bs[cur], nn, lane);
        acc[ms][nn] = __builtin_amdgcn_wmma_f32_16x16x32_f16(false, a, false, bf, (short)0,
                                                             acc[ms][nn], false, false);
      }
    }
  }
}

__global__ __launch_bounds__(BLOCK)
void condconv1_kernel(const float* __restrict__ x, const _Float16* __restrict__ cw,
                      const float* __restrict__ g, const float* __restrict__ bb,
                      const float* __restrict__ mu, const float* __restrict__ var,
                      float* __restrict__ out) {
  __shared__ __align__(16) _Float16 As[2][TILE_M * LDS_STRIDE];
  __shared__ __align__(16) _Float16 Bs[2][TILE_N * LDS_STRIDE];
  const int tid = threadIdx.x, lane = tid & 31, wave = tid >> 5;
  const int b = blockIdx.z, obase = blockIdx.y * TILE_M, pbase = blockIdx.x * TILE_N;
  v8f acc[2][4];
  v8f zero = {};
  #pragma unroll
  for (int m = 0; m < 2; ++m)
    #pragma unroll
    for (int n = 0; n < 4; ++n) acc[m][n] = zero;
  gemm1x1_body(x + (size_t)b * INPC * PIX, INPC,
               cw + ((size_t)b * WIDTHC + obase) * INPC, As, Bs, acc, tid, wave, lane, pbase);
  store_bn(acc, obase, pbase, wave, lane, g, bb, mu, var, nullptr,
           out + (size_t)b * WIDTHC * PIX);
}

__global__ __launch_bounds__(BLOCK)
void condconv3_kernel(const float* __restrict__ in, const _Float16* __restrict__ cw,
                      const float* __restrict__ g, const float* __restrict__ bb,
                      const float* __restrict__ mu, const float* __restrict__ var,
                      const float* __restrict__ resid, float* __restrict__ out) {
  __shared__ __align__(16) _Float16 As[2][TILE_M * LDS_STRIDE];
  __shared__ __align__(16) _Float16 Bs[2][TILE_N * LDS_STRIDE];
  const int tid = threadIdx.x, lane = tid & 31, wave = tid >> 5;
  const int b = blockIdx.z, obase = blockIdx.y * TILE_M, pbase = blockIdx.x * TILE_N;
  v8f acc[2][4];
  v8f zero = {};
  #pragma unroll
  for (int m = 0; m < 2; ++m)
    #pragma unroll
    for (int n = 0; n < 4; ++n) acc[m][n] = zero;
  gemm1x1_body(in + (size_t)b * WIDTHC * PIX, WIDTHC,
               cw + ((size_t)b * OUTPC + obase) * WIDTHC, As, Bs, acc, tid, wave, lane, pbase);
  store_bn(acc, obase, pbase, wave, lane, g, bb, mu, var,
           resid + (size_t)b * OUTPC * PIX, out + (size_t)b * OUTPC * PIX);
}

// ---- stage 2: 3x3 pad1, implicit im2col -----------------------------------
__global__ __launch_bounds__(BLOCK)
void condconv2_kernel(const float* __restrict__ in, const _Float16* __restrict__ cw,
                      const float* __restrict__ g, const float* __restrict__ bb,
                      const float* __restrict__ mu, const float* __restrict__ var,
                      float* __restrict__ out) {
  __shared__ __align__(16) _Float16 As[2][TILE_M * LDS_STRIDE];
  __shared__ __align__(16) _Float16 Bs[2][TILE_N * LDS_STRIDE];
  const int tid = threadIdx.x, lane = tid & 31, wave = tid >> 5;
  const int b = blockIdx.z, obase = blockIdx.y * TILE_M, pbase = blockIdx.x * TILE_N;
  const int KTOT = WIDTHC * 9;
  const int NIT = KTOT / TILE_K;

  v8f acc[2][4];
  v8f zero = {};
  #pragma unroll
  for (int m = 0; m < 2; ++m)
    #pragma unroll
    for (int n = 0; n < 4; ++n) acc[m][n] = zero;

  const float* inb = in + (size_t)b * WIDTHC * PIX;
  const _Float16* cwm = cw + ((size_t)b * WIDTHC + obase) * KTOT;

  const int n = tid & 63, kseg = (tid >> 6) * 16;
  const int p = pbase + n;
  const int y = p / HH, xx = p - y * HH;

  // per-tap column staging: fixed pixel -> fixed shifted source offset + flag
  auto stageB2 = [&](int kb, _Float16* Bsb) {
    const int tap = kb / WIDTHC;          // uniform: 256 % 32 == 0
    const int ibase = kb % WIDTHC;
    const int sy = y + tap / 3 - 1, sx = xx + tap % 3 - 1;
    const bool ok = (p < PIX) & (sy >= 0) & (sy < HH) & (sx >= 0) & (sx < HH);
    const int off = ok ? (sy * HH + sx) : 0;
    stage_B_cols(inb + (size_t)ibase * PIX, off, ok, n, kseg, Bsb);
  };

  stage_A_issue(cwm, KTOT, 0, As[0], tid, wave);
  stageB2(0, Bs[0]);

  for (int it = 0; it < NIT; ++it) {
    const int cur = it & 1;
    stage_A_wait(wave);
    __syncthreads();
    if (it + 1 < NIT) {
      const int kn = (it + 1) * TILE_K;
      stage_A_issue(cwm, KTOT, kn, As[cur ^ 1], tid, wave);
      stageB2(kn, Bs[cur ^ 1]);
    }
    #pragma unroll
    for (int ms = 0; ms < 2; ++ms) {
      v16h a = load_a_frag(As[cur], wave * 2 + ms, lane);
      #pragma unroll
      for (int nn = 0; nn < 4; ++nn) {
        v16h bf = load_b_frag(Bs[cur], nn, lane);
        acc[ms][nn] = __builtin_amdgcn_wmma_f32_16x16x32_f16(false, a, false, bf, (short)0,
                                                             acc[ms][nn], false, false);
      }
    }
  }
  store_bn(acc, obase, pbase, wave, lane, g, bb, mu, var, nullptr,
           out + (size_t)b * WIDTHC * PIX);
}

// ---- launch ---------------------------------------------------------------
extern "C" void kernel_launch(void* const* d_in, const int* in_sizes, int n_in,
                              void* d_out, int out_size, void* d_ws, size_t ws_size,
                              hipStream_t stream) {
  const float* x    = (const float*)d_in[0];
  const float* w1   = (const float*)d_in[1];
  const float* w2   = (const float*)d_in[2];
  const float* w3   = (const float*)d_in[3];
  const float* r1_w = (const float*)d_in[4];
  const float* r1_b = (const float*)d_in[5];
  const float* r2_w = (const float*)d_in[6];
  const float* r2_b = (const float*)d_in[7];
  const float* r3_w = (const float*)d_in[8];
  const float* r3_b = (const float*)d_in[9];
  const float* bn1_g = (const float*)d_in[10], *bn1_b = (const float*)d_in[11];
  const float* bn1_m = (const float*)d_in[12], *bn1_v = (const float*)d_in[13];
  const float* bn2_g = (const float*)d_in[14], *bn2_b = (const float*)d_in[15];
  const float* bn2_m = (const float*)d_in[16], *bn2_v = (const float*)d_in[17];
  const float* bn3_g = (const float*)d_in[18], *bn3_b = (const float*)d_in[19];
  const float* bn3_m = (const float*)d_in[20], *bn3_v = (const float*)d_in[21];

  float* ws      = (float*)d_ws;
  float* pooled1 = ws;                                   // 32*1024
  float* pooled2 = pooled1 + BATCH * INPC;               // 32*256
  float* pooled3 = pooled2 + BATCH * WIDTHC;             // 32*256
  float* rw1     = pooled3 + BATCH * WIDTHC;             // 256
  float* rw2     = rw1 + BATCH * NEXP;
  float* rw3     = rw2 + BATCH * NEXP;
  float* out1    = rw3 + BATCH * NEXP;                   // 32*256*196
  float* out2    = out1 + (size_t)BATCH * WIDTHC * PIX;  // 32*256*196
  _Float16* cwh  = (_Float16*)(out2 + (size_t)BATCH * WIDTHC * PIX); // reused per stage
  float* outp    = (float*)d_out;

  const dim3 blk(BLOCK);

  // stage 1
  pool_kernel<<<dim3((BATCH * INPC + 255) / 256), 256, 0, stream>>>(x, pooled1, INPC);
  route_kernel<<<1, 256, 0, stream>>>(pooled1, r1_w, r1_b, rw1, INPC);
  combine13_kernel<<<dim3(WIDTHC * INPC / 8 / 256, BATCH), 256, 0, stream>>>(w1, rw1, cwh, WIDTHC, INPC);
  condconv1_kernel<<<dim3(4, WIDTHC / TILE_M, BATCH), blk, 0, stream>>>(
      x, cwh, bn1_g, bn1_b, bn1_m, bn1_v, out1);

  // stage 2
  pool_kernel<<<dim3((BATCH * WIDTHC + 255) / 256), 256, 0, stream>>>(out1, pooled2, WIDTHC);
  route_kernel<<<1, 256, 0, stream>>>(pooled2, r2_w, r2_b, rw2, WIDTHC);
  combine2_kernel<<<dim3(WIDTHC * WIDTHC * 9 / 8 / 256, BATCH), 256, 0, stream>>>(w2, rw2, cwh);
  condconv2_kernel<<<dim3(4, WIDTHC / TILE_M, BATCH), blk, 0, stream>>>(
      out1, cwh, bn2_g, bn2_b, bn2_m, bn2_v, out2);

  // stage 3
  pool_kernel<<<dim3((BATCH * WIDTHC + 255) / 256), 256, 0, stream>>>(out2, pooled3, WIDTHC);
  route_kernel<<<1, 256, 0, stream>>>(pooled3, r3_w, r3_b, rw3, WIDTHC);
  combine13_kernel<<<dim3(OUTPC * WIDTHC / 8 / 256, BATCH), 256, 0, stream>>>(w3, rw3, cwh, OUTPC, WIDTHC);
  condconv3_kernel<<<dim3(4, OUTPC / TILE_M, BATCH), blk, 0, stream>>>(
      out2, cwh, bn3_g, bn3_b, bn3_m, bn3_v, x, outp);
}
